// CharRNN_90348932038680
// MI455X (gfx1250) — matrix operand order repaired and checked by reading, compiled, not run on hardware
//
#include <hip/hip_runtime.h>
#include <hip/hip_bf16.h>

// ---------------------------------------------------------------------------
// 3x stacked shared-weight LSTM (B=50, T=2048, D=65) + Dense(65) for gfx1250.
//   prep_kernel    : pack [W;U] (130->160 x 260->272) and Wd (65x65 -> 96x80)
//                    into per-lane WMMA bf16 B-fragment layouts in workspace.
//   lstm_seq_kernel: ONE persistent workgroup (32 waves) on one WGP; weights,
//                    activations, h/c state LDS-resident. Per (t,layer):
//                    gemm (68 16x16 tiles, 5x v_wmma_f32_16x16x32_bf16 each)
//                    -> barrier -> gates (fp32, native v_tanh) which also
//                    stages the NEXT layer's A matrix -> barrier.
//                    Only 6 barriers per timestep on the sequential chain.
//   dense_kernel   : massively parallel Y(bf16) @ Wd + bd, 32000 warp-tiles.
// ---------------------------------------------------------------------------

#define T_SEQ   2048
#define BATCH   50
#define UNITS_  65
#define GATES   260           // 4*UNITS
#define KCAT    160           // concat K: 65(inp)+65(h)+30 zero pad
#define MT_Z    4             // 64 rows  / 16
#define NT_Z    17            // 272 cols / 16
#define KT_Z    5             // 160      / 32
#define YSTRIDE 66            // bf16 row stride for Y (4B-aligned pair loads)
#define WFRAG_ELEMS  (KT_Z * NT_Z * 512)   // 43520 bf16
#define WDFRAG_ELEMS (3 * 5 * 512)         // 7680  bf16

typedef __attribute__((ext_vector_type(16))) __bf16 bf16x16;
typedef __attribute__((ext_vector_type(8)))  float  f32x8;

static __device__ __forceinline__ unsigned short f2bf(float f) {
    unsigned u = __float_as_uint(f);
    u += 0x7FFFu + ((u >> 16) & 1u);          // round-to-nearest-even
    return (unsigned short)(u >> 16);
}
static __device__ __forceinline__ float sigf(float x) {
    return 1.0f / (1.0f + __expf(-x));        // v_exp_f32 path
}
static __device__ __forceinline__ float fast_tanh(float x) {
#if __has_builtin(__builtin_amdgcn_tanhf)
    return __builtin_amdgcn_tanhf(x);         // native TRANS32 v_tanh_f32
#else
    return tanhf(x);
#endif
}

// --- pack weights into WMMA B-fragment layouts (run once per launch) --------
// B-matrix (32x16 KxN) mapping: lane = n_local + 16*(k_local/16), elem = k_local%16
__global__ void prep_kernel(const float* __restrict__ W, const float* __restrict__ U,
                            const float* __restrict__ Wd,
                            unsigned short* __restrict__ wfrag,
                            unsigned short* __restrict__ wdfrag) {
    int idx = blockIdx.x * blockDim.x + threadIdx.x;
    if (idx < WFRAG_ELEMS) {
        int frag = idx >> 9, within = idx & 511;
        int kt = frag / NT_Z, nt = frag % NT_Z;
        int lane = within >> 4, e = within & 15;
        int k = kt * 32 + (lane >> 4) * 16 + e;
        int n = nt * 16 + (lane & 15);
        float v = 0.f;
        if (n < GATES) {
            if (k < UNITS_)           v = W[k * GATES + n];
            else if (k < 2 * UNITS_)  v = U[(k - UNITS_) * GATES + n];
        }
        wfrag[idx] = f2bf(v);
    } else {
        int idx2 = idx - WFRAG_ELEMS;
        if (idx2 < WDFRAG_ELEMS) {
            int frag = idx2 >> 9, within = idx2 & 511;
            int kt = frag / 5, nt = frag % 5;
            int lane = within >> 4, e = within & 15;
            int k = kt * 32 + (lane >> 4) * 16 + e;
            int n = nt * 16 + (lane & 15);
            float v = (k < UNITS_ && n < UNITS_) ? Wd[k * UNITS_ + n] : 0.f;
            wdfrag[idx2] = f2bf(v);
        }
    }
}

// --- persistent sequential LSTM on a single WGP -----------------------------
__global__ __launch_bounds__(1024) void lstm_seq_kernel(
    const float* __restrict__ x, const float* __restrict__ b,
    const unsigned short* __restrict__ wfrag_g,
    unsigned short* __restrict__ Y)
{
    __shared__ __align__(16) unsigned short sWfrag[WFRAG_ELEMS];     //  87.0 KB
    __shared__ __align__(16) unsigned short sA[64 * KCAT];           //  20.5 KB
    __shared__ float          sZ[BATCH * GATES];                     //  52.0 KB
    __shared__ float          sC[3][BATCH * UNITS_];                 //  39.0 KB
    __shared__ __align__(4) unsigned short sH[3][BATCH * UNITS_];    //  19.5 KB
    __shared__ float          sBias[GATES];                          //   1.0 KB

    const int tid  = threadIdx.x;
    const int lane = tid & 31;
    const int wave = tid >> 5;                 // 0..31

    // One-time init (re-done every launch: deterministic under graph replay).
    for (int i = tid; i < WFRAG_ELEMS / 8; i += 1024)
        ((uint4*)sWfrag)[i] = ((const uint4*)wfrag_g)[i];
    for (int i = tid; i < 64 * KCAT; i += 1024) sA[i] = 0;   // pads stay zero
    for (int i = tid; i < BATCH * UNITS_; i += 1024) {
        sC[0][i] = 0.f; sC[1][i] = 0.f; sC[2][i] = 0.f;
        sH[0][i] = 0;   sH[1][i] = 0;   sH[2][i] = 0;
    }
    for (int i = tid; i < GATES; i += 1024) sBias[i] = b[i];
    // stage A for (t=0, layer=0): inp = x[:,0,:], h part stays zero
    for (int e = tid; e < BATCH * UNITS_; e += 1024) {
        int m = e / UNITS_, n = e - m * UNITS_;
        sA[m * KCAT + n] = f2bf(x[(m * T_SEQ) * UNITS_ + n]);
    }
    if (tid < 250) {
        int m = tid / 5, seg = tid % 5;
        __builtin_prefetch(&x[(m * T_SEQ + 1) * UNITS_ + seg * 13], 0, 0);
    }
    __syncthreads();

    for (int t = 0; t < T_SEQ; ++t) {
        for (int l = 0; l < 3; ++l) {
            // --- z = A x [W;U]: 4x17 tiles over 32 waves, 5 WMMAs each -----
            for (int tile = wave; tile < MT_Z * NT_Z; tile += 32) {
                int mt = tile / NT_Z, nt = tile % NT_Z;
                f32x8 acc = {};
                const int mrow = mt * 16 + (lane & 15);   // A: m = lane%16
                const int b8   = lane >> 4;               // A: (k/16)%2 bit
                #pragma unroll
                for (int kt = 0; kt < KT_Z; ++kt) {
                    union { bf16x16 v; unsigned u[8]; } a;
                    #pragma unroll
                    for (int p = 0; p < 8; ++p) {         // coalesces to 2x ds_load_b128
                        int kl = kt * 32 + b8 * 16 + p * 2;
                        a.u[p] = *(const unsigned*)&sA[mrow * KCAT + kl];
                    }
                    const bf16x16 bm =
                        *(const bf16x16*)&sWfrag[(kt * NT_Z + nt) * 512 + lane * 16];
                    acc = __builtin_amdgcn_wmma_f32_16x16x32_bf16(
                              false, a.v, false, bm, (short)0, acc, false, false);
                }
                // store z tile; guards are wave-uniform except the edge tiles
                const int n     = nt * 16 + (lane & 15);
                const bool nOK  = (nt < 16) | (n < GATES);      // per-lane only for nt==16
                float* zp = &sZ[(mt * 16 + 8 * (lane >> 4)) * GATES + n];
                if (mt < 3) {                                   // rows all < 50
                    if (nOK) {
                        #pragma unroll
                        for (int v = 0; v < 8; ++v) zp[v * GATES] = acc[v];
                    }
                } else {                                        // mt==3: rows 48..63
                    if ((lane >> 4) == 0 && nOK) {              // only m=48,49 live
                        zp[0] = acc[0];
                        zp[GATES] = acc[1];
                    }
                }
            }
            __syncthreads();

            // --- gates (fp32) + stage NEXT A matrix ------------------------
            for (int e = tid; e < BATCH * UNITS_; e += 1024) {
                int m = e / UNITS_, n = e - m * UNITS_;
                const float* zr = &sZ[m * GATES];
                float zi = zr[n]              + sBias[n];
                float zf = zr[n + UNITS_]     + sBias[n + UNITS_];
                float zg = zr[n + 2 * UNITS_] + sBias[n + 2 * UNITS_];
                float zo = zr[n + 3 * UNITS_] + sBias[n + 3 * UNITS_];
                float cn = sigf(zf) * sC[l][e] + sigf(zi) * fast_tanh(zg);
                float hn = sigf(zo) * fast_tanh(cn);
                sC[l][e] = cn;
                unsigned short hb = f2bf(hn);
                sH[l][e] = hb;
                if (l < 2) {
                    // next layer's input = h_l, next layer's recurrent = sH[l+1]
                    sA[m * KCAT + n]          = hb;
                    sA[m * KCAT + UNITS_ + n] = sH[l + 1][e];
                } else {
                    Y[(t * BATCH + m) * YSTRIDE + n] = hb;
                    if (t + 1 < T_SEQ) {      // stage (t+1, layer 0)
                        sA[m * KCAT + n]          = f2bf(x[(m * T_SEQ + t + 1) * UNITS_ + n]);
                        sA[m * KCAT + UNITS_ + n] = sH[0][e];
                    }
                }
            }
            if (l == 2) {
                if (tid < BATCH)              // keep pad column zero for dense
                    Y[(t * BATCH + tid) * YSTRIDE + UNITS_] = 0;
                if (t + 2 < T_SEQ && tid < 250) {   // hide x latency
                    int m = tid / 5, seg = tid % 5;
                    __builtin_prefetch(&x[(m * T_SEQ + t + 2) * UNITS_ + seg * 13], 0, 0);
                }
            }
            __syncthreads();
        }
    }
}

// --- out = Y @ Wd + bd : fully parallel, all WGPs ---------------------------
__global__ __launch_bounds__(256) void dense_kernel(
    const unsigned short* __restrict__ Y,
    const unsigned short* __restrict__ wdfrag,
    const float* __restrict__ bd,
    float* __restrict__ out)
{
    const int lane = threadIdx.x & 31;
    const int gw   = blockIdx.x * 8 + (threadIdx.x >> 5);   // 0..31999 warp-tiles
    const int mt   = gw / 5, nt = gw % 5;
    const int mlo  = lane & 15;
    const int b8   = lane >> 4;

    f32x8 acc = {};
    #pragma unroll
    for (int kt = 0; kt < 3; ++kt) {
        union { bf16x16 v; unsigned u[8]; } a;
        #pragma unroll
        for (int p = 0; p < 8; ++p) {
            int k = kt * 32 + b8 * 16 + p * 2;
            a.u[p] = (k < UNITS_)
                ? *(const unsigned*)&Y[(mt * 16 + mlo) * YSTRIDE + k]  // k=64 pairs with zeroed col 65
                : 0u;
        }
        const bf16x16 bm = *(const bf16x16*)&wdfrag[(kt * 5 + nt) * 512 + lane * 16];
        acc = __builtin_amdgcn_wmma_f32_16x16x32_bf16(
                  false, a.v, false, bm, (short)0, acc, false, false);
    }
    const int n = nt * 16 + mlo;
    if (n < UNITS_) {
        float bdv = bd[n];
        #pragma unroll
        for (int v = 0; v < 8; ++v) {
            int r  = mt * 16 + v + 8 * b8;          // Y row = t*BATCH + b
            int tt = r / BATCH, bi = r - tt * BATCH;
            out[(bi * T_SEQ + tt) * UNITS_ + n] = acc[v] + bdv;
        }
    }
}

extern "C" void kernel_launch(void* const* d_in, const int* in_sizes, int n_in,
                              void* d_out, int out_size, void* d_ws, size_t ws_size,
                              hipStream_t stream) {
    const float* x  = (const float*)d_in[0];
    const float* W  = (const float*)d_in[1];
    const float* U  = (const float*)d_in[2];
    const float* b  = (const float*)d_in[3];
    const float* Wd = (const float*)d_in[4];
    const float* bd = (const float*)d_in[5];
    float* out = (float*)d_out;

    // workspace layout (bf16): [Wcat frags 43520][Wd frags 7680][Y 102400*66]
    unsigned short* wfrag  = (unsigned short*)d_ws;
    unsigned short* wdfrag = wfrag + WFRAG_ELEMS;
    unsigned short* Y      = wfrag + WFRAG_ELEMS + WDFRAG_ELEMS;  // 4B-aligned

    prep_kernel<<<200, 256, 0, stream>>>(W, U, Wd, wfrag, wdfrag);
    lstm_seq_kernel<<<1, 1024, 0, stream>>>(x, b, wfrag, Y);
    dense_kernel<<<4000, 256, 0, stream>>>(Y, wdfrag, bd, out);
}
